// attention_block_41918880809173
// MI455X (gfx1250) — matrix operand order repaired and checked by reading, not compiled, not run on hardware
//
#include <hip/hip_runtime.h>

// ---------------------------------------------------------------------------
// MI455X (gfx1250, wave32) cross-attention block.
// Heavy math via v_wmma_f32_16x16x32_bf16; tile staging via
// global_load_async_to_lds_b128 (ASYNCcnt) where the data is already bf16.
// ---------------------------------------------------------------------------

typedef __bf16 bf16_t;
typedef __attribute__((ext_vector_type(16))) __bf16 v16bf;
typedef __attribute__((ext_vector_type(8)))  __bf16 v8bf;
typedef __attribute__((ext_vector_type(8)))  float  v8f;

union Frag {
  v16bf  v;
  v8bf   h[2];
  __bf16 e[16];
};

__device__ __forceinline__ v8f wmma_bf16(v16bf a, v16bf b, v8f c) {
  // D = A(16x32) * B(32x16) + C, f32 accumulate
  return __builtin_amdgcn_wmma_f32_16x16x32_bf16(
      /*neg_a=*/false, a, /*neg_b=*/false, b,
      /*c_mod=*/(short)0, c, /*reuse_a=*/false, /*reuse_b=*/false);
}

// Async global->LDS copy of 16 bytes per lane (CDNA5, tracked on ASYNCcnt).
__device__ __forceinline__ void async_b128(unsigned lds_off, const void* gaddr) {
  asm volatile("global_load_async_to_lds_b128 %0, %1, off"
               :: "v"(lds_off), "v"(gaddr) : "memory");
}
__device__ __forceinline__ void wait_async0() {
  asm volatile("s_wait_asynccnt 0x0" ::: "memory");
}
__device__ __forceinline__ unsigned lds_addr(const void* p) {
  return (unsigned)(uintptr_t)p;
}

#define C_DIM  256
#define HEADS  4
#define HDIM   64
#define BATCH  4
#define NSP    2048
#define BN_EPS 1e-5f

// ---------------------------------------------------------------------------
// fp32 (B,C,N) -> bf16 (B,N,C)  (channel-last for WMMA A-operands)
// ---------------------------------------------------------------------------
__global__ __launch_bounds__(256) void k_to_cl_bf16(
    const float* __restrict__ x, bf16_t* __restrict__ y) {
  size_t i = (size_t)blockIdx.x * 256 + threadIdx.x;  // over B*C*N
  int    n = (int)(i % NSP);
  size_t t = i / NSP;
  int    c = (int)(t % C_DIM);
  int    b = (int)(t / C_DIM);
  y[((size_t)b * NSP + n) * C_DIM + c] = (bf16_t)x[i];
}

// Elementwise fp32 -> bf16 (weight pre-convert so GEMM tiles can async-copy)
__global__ __launch_bounds__(256) void k_f32_to_bf16(
    const float* __restrict__ x, bf16_t* __restrict__ y, int n) {
  int i = blockIdx.x * 256 + threadIdx.x;
  if (i < n) y[i] = (bf16_t)x[i];
}

// ---------------------------------------------------------------------------
// Generic channel-last GEMM:  out[o, n] = sum_c W[o, c] * X[b][n][c] + bias[o]
// Block: 128 thr = 4 waves (2n x 2o); WG tile 128n x 128o; wave tile 64n x 64o
// -> 16 WMMA per K-step vs 8 global + 8 LDS fragment loads.
// Weight tile staged via async global->LDS copies (bf16 weights).
// MODE 0: store bf16 channel-last (B,N,O)
// MODE 1: BN (inference) + ReLU, store bf16 channel-last
// MODE 2: residual add with fp32 (B,O,N), store fp32 (B,O,N)   [final output]
// MODE 3: Q/K scatter -> (B,H,N,64) bf16   (c = d*4 + h)
// MODE 4: V   scatter -> (B,H,64,N) bf16
// Concat: c < Csplit reads X0 (row stride Csplit), else X1 (stride Cin-Csplit)
// ---------------------------------------------------------------------------
template <int MODE>
__global__ __launch_bounds__(128) void k_gemm_cl(
    const bf16_t* __restrict__ X0, const bf16_t* __restrict__ X1,
    const bf16_t* __restrict__ W, const float* __restrict__ bias,
    const float* __restrict__ bng, const float* __restrict__ bnb,
    const float* __restrict__ bnm, const float* __restrict__ bnv,
    const float* __restrict__ resid, void* __restrict__ Out,
    int Cin, int Csplit, int O, int Nsp) {
  __shared__ bf16_t Wt[128][40];  // padded row stride (80 B, 16B aligned)

  const int tid  = threadIdx.x;
  const int wave = tid >> 5;
  const int lane = tid & 31;
  const int l16  = lane & 15;
  const int hi   = lane >> 4;
  const int wn   = wave & 1;   // 2 waves along n
  const int wo   = wave >> 1;  // 2 waves along o
  const int b    = blockIdx.z;
  const int n0   = blockIdx.x * 128 + wn * 64;
  const int o0   = blockIdx.y * 128 + wo * 64;

  v8f acc[4][4] = {};

  const int ksteps = Cin >> 5;
  for (int ks = 0; ks < ksteps; ++ks) {
    const int c0 = ks << 5;
    __syncthreads();  // previous-iteration Wt reads complete
    {                 // stage 128x32 bf16 weight tile: 1 row / thread, 4 asyncs
      const bf16_t* wp = W + (size_t)(blockIdx.y * 128 + tid) * Cin + c0;
#pragma unroll
      for (int s = 0; s < 4; ++s)
        async_b128(lds_addr(&Wt[tid][s * 8]), wp + s * 8);
    }
    wait_async0();
    __syncthreads();

    // ---- A fragments straight from global (L2-resident, channel-last) ----
    const bf16_t* xbase;
    int cs, cc;
    if (c0 < Csplit) { xbase = X0; cs = Csplit;       cc = c0; }
    else             { xbase = X1; cs = Cin - Csplit; cc = c0 - Csplit; }
    Frag a[4];
#pragma unroll
    for (int i = 0; i < 4; ++i) {
      const int     n = n0 + i * 16 + l16;
      const bf16_t* p = xbase + ((size_t)b * Nsp + n) * cs + cc + hi * 8;
      a[i].h[0] = *(const v8bf*)(p);       // K {0..7} / {8..15}
      a[i].h[1] = *(const v8bf*)(p + 16);  // K {16..23} / {24..31}
    }

    // ---- B fragments from LDS + 16 WMMAs ----
#pragma unroll
    for (int j = 0; j < 4; ++j) {
      Frag bw;
      const bf16_t* p = &Wt[wo * 64 + j * 16 + l16][hi * 8];
      bw.h[0] = *(const v8bf*)(p);
      bw.h[1] = *(const v8bf*)(p + 16);
#pragma unroll
      for (int i = 0; i < 4; ++i) acc[i][j] = wmma_bf16(a[i].v, bw.v, acc[i][j]);
    }
  }

  // ---- epilogue: accumulator lane<->o, VGPR<->n ----
#pragma unroll
  for (int j = 0; j < 4; ++j) {
    const int   o  = o0 + j * 16 + l16;
    const float bs = bias[o];
    float sc = 0.f, sh = 0.f;
    if (MODE == 1) {
      sc = bng[o] * __frsqrt_rn(bnv[o] + BN_EPS);
      sh = bnb[o] - bnm[o] * sc;
    }
#pragma unroll
    for (int i = 0; i < 4; ++i) {
#pragma unroll
      for (int r = 0; r < 8; ++r) {
        const int n = n0 + i * 16 + r + hi * 8;
        float v = acc[i][j][r] + bs;
        if (MODE == 0) {
          ((bf16_t*)Out)[((size_t)b * Nsp + n) * O + o] = (bf16_t)v;
        } else if (MODE == 1) {
          v = fmaxf(v * sc + sh, 0.f);
          ((bf16_t*)Out)[((size_t)b * Nsp + n) * O + o] = (bf16_t)v;
        } else if (MODE == 2) {
          const size_t idx = ((size_t)b * O + o) * Nsp + n;
          ((float*)Out)[idx] = v + resid[idx];
        } else if (MODE == 3) {  // Q/K: (B,H,N,64), channel o = d*4 + h
          const int d = o >> 2, h = o & 3;
          ((bf16_t*)Out)[(((size_t)b * HEADS + h) * Nsp + n) * HDIM + d] = (bf16_t)v;
        } else {                 // V: (B,H,64,N)
          const int d = o >> 2, h = o & 3;
          ((bf16_t*)Out)[(((size_t)b * HEADS + h) * HDIM + d) * Nsp + n] = (bf16_t)v;
        }
      }
    }
  }
}

// ---------------------------------------------------------------------------
// Flash cross-attention, one (b,h) per blockIdx.y, 64 q-rows per block.
// Q,K: (B,H,N,64) bf16;  V: (B,H,64,N) bf16;  av out: (B,N,256) bf16, c=d*4+h
// 128 threads = 4 waves, each wave owns a 16-row q tile.
// S^T WMMA: A = K tile (m x d), B = Q (per-lane fixed n, contiguous d).
// Accumulator P layout == A-fragment layout -> chained PV WMMA for free.
// K/V tiles staged via async global->LDS copies.
// ---------------------------------------------------------------------------
__global__ __launch_bounds__(128) void k_flash_attn(
    const bf16_t* __restrict__ Q, const bf16_t* __restrict__ K,
    const bf16_t* __restrict__ V, bf16_t* __restrict__ av, int Nq, int Nk) {
  __shared__ bf16_t Kt[32][72];  // m x d   (144 B rows)
  __shared__ bf16_t Vt[64][40];  // d x m   (80 B rows)
  __shared__ float  red[4][16];

  const int tid  = threadIdx.x;
  const int wave = tid >> 5;
  const int lane = tid & 31;
  const int l16  = lane & 15;
  const int hi   = lane >> 4;
  const int bh   = blockIdx.y;
  const int b    = bh >> 2;
  const int h    = bh & 3;
  const int qb   = blockIdx.x * 64 + wave * 16;

  const bf16_t* Qp = Q + (size_t)bh * Nq * HDIM;
  const bf16_t* Kp = K + (size_t)bh * Nk * HDIM;
  const bf16_t* Vp = V + (size_t)bh * HDIM * Nk;

  // Q B-fragments (held in VGPRs for the whole kernel): lane -> n = qb+l16
  Frag qf[2];
  {
    const int n = qb + l16;
#pragma unroll
    for (int kc = 0; kc < 2; ++kc) {
      const bf16_t* p = Qp + (size_t)n * HDIM + kc * 32 + hi * 8;
      qf[kc].h[0] = *(const v8bf*)(p);
      qf[kc].h[1] = *(const v8bf*)(p + 16);
    }
  }

  v8f   oacc[4] = {};             // O^T tiles: lane<->d, VGPR<->n
  float mx = -3.0e38f, lsum = 0.f;
  const float log2e_scale = 0.125f * 1.44269504088896f;  // 1/sqrt(64) * log2(e)

  const int kr = tid >> 2, ksg = (tid & 3) << 4;  // K tile: 32 x 64
  const int vr = tid >> 1, vsg = (tid & 1) << 4;  // V tile: 64 x 32

  for (int m0 = 0; m0 < Nk; m0 += 32) {
    __syncthreads();  // previous-iteration tile reads complete
    {                 // stage tiles with async global->LDS copies
      const bf16_t* kp = Kp + (size_t)(m0 + kr) * HDIM + ksg;
      async_b128(lds_addr(&Kt[kr][ksg]), kp);
      async_b128(lds_addr(&Kt[kr][ksg + 8]), kp + 8);
      const bf16_t* vp = Vp + (size_t)vr * Nk + m0 + vsg;
      async_b128(lds_addr(&Vt[vr][vsg]), vp);
      async_b128(lds_addr(&Vt[vr][vsg + 8]), vp + 8);
    }
    wait_async0();
    __syncthreads();

    // ---- S^T = K * Q : two m-tiles, K-dim = 64 in two chunks ----
    v8f sacc[2] = {};
#pragma unroll
    for (int mt = 0; mt < 2; ++mt) {
#pragma unroll
      for (int kc = 0; kc < 2; ++kc) {
        Frag kf;  // A operand: lane row m, per-lane d in {0..7,16..23}/{8..15,24..31}
        const bf16_t* p = &Kt[mt * 16 + l16][kc * 32 + hi * 8];
        kf.h[0] = *(const v8bf*)(p);
        kf.h[1] = *(const v8bf*)(p + 16);
        sacc[mt] = wmma_bf16(kf.v, qf[kc].v, sacc[mt]);
      }
    }

    // ---- online softmax: per lane fixed n, 16 m-values across VGPRs ----
    float s[16];
#pragma unroll
    for (int i = 0; i < 8; ++i) { s[i] = sacc[0][i]; s[8 + i] = sacc[1][i]; }
    float ml = s[0];
#pragma unroll
    for (int i = 1; i < 16; ++i) ml = fmaxf(ml, s[i]);
    ml = fmaxf(ml, __shfl_xor(ml, 16, 32));  // combine lane L <-> L+16 halves
    const float mnew   = fmaxf(mx, ml);
    const float rscale = exp2f((mx - mnew) * log2e_scale);
    mx = mnew;
    float ps = 0.f;
#pragma unroll
    for (int i = 0; i < 16; ++i) {
      s[i] = exp2f((s[i] - mnew) * log2e_scale);
      ps += s[i];
    }
    ps += __shfl_xor(ps, 16, 32);
    lsum = lsum * rscale + ps;

    // broadcast per-n rescale into O^T layout (lane<->d needs r[n] per VGPR)
    if (hi == 0) red[wave][l16] = rscale;
    asm volatile("s_wait_dscnt 0" ::: "memory");
    float rv[8];
#pragma unroll
    for (int r = 0; r < 8; ++r) rv[r] = red[wave][r + hi * 8];
#pragma unroll
    for (int j = 0; j < 4; ++j)
#pragma unroll
      for (int r = 0; r < 8; ++r) oacc[j][r] *= rv[r];

    // ---- PV: P accumulator doubles as A fragment ----
    Frag pf;
#pragma unroll
    for (int i = 0; i < 16; ++i) pf.e[i] = (bf16_t)s[i];
#pragma unroll
    for (int j = 0; j < 4; ++j) {
      Frag vf;  // B operand: lane col d, per-lane m in {0..7,16..23}/{8..15,24..31}
      const bf16_t* p = &Vt[j * 16 + l16][hi * 8];
      vf.h[0] = *(const v8bf*)(p);
      vf.h[1] = *(const v8bf*)(p + 16);
      oacc[j] = wmma_bf16(pf.v, vf.v, oacc[j]);
    }
  }

  // ---- finalize: divide by row sums, scatter to (B,N,256), c = d*4+h ----
  if (hi == 0) red[wave][l16] = 1.0f / lsum;
  asm volatile("s_wait_dscnt 0" ::: "memory");
  float linv[8];
#pragma unroll
  for (int r = 0; r < 8; ++r) linv[r] = red[wave][r + hi * 8];

  bf16_t* avp = av + (size_t)b * Nq * C_DIM;
#pragma unroll
  for (int j = 0; j < 4; ++j) {
    const int d = j * 16 + l16;
    const int c = d * HEADS + h;
#pragma unroll
    for (int r = 0; r < 8; ++r) {
      const int n = qb + r + hi * 8;
      avp[(size_t)n * C_DIM + c] = (bf16_t)(oacc[j][r] * linv[r]);
    }
  }
}

// ---------------------------------------------------------------------------
// Host-side orchestration
// ---------------------------------------------------------------------------
extern "C" void kernel_launch(void* const* d_in, const int* in_sizes, int n_in,
                              void* d_out, int out_size, void* d_ws, size_t ws_size,
                              hipStream_t stream) {
  (void)in_sizes; (void)n_in; (void)out_size; (void)ws_size;
  const float* fea1 = (const float*)d_in[0];
  const float* fea2 = (const float*)d_in[1];
  const float* wq   = (const float*)d_in[2];
  const float* bq   = (const float*)d_in[3];
  const float* wk   = (const float*)d_in[4];
  const float* bk   = (const float*)d_in[5];
  const float* wv   = (const float*)d_in[6];
  const float* bv   = (const float*)d_in[7];
  const float* wm   = (const float*)d_in[8];
  const float* bm   = (const float*)d_in[9];
  const float* wa1  = (const float*)d_in[10];
  const float* ba1  = (const float*)d_in[11];
  const float* bng  = (const float*)d_in[12];
  const float* bnb  = (const float*)d_in[13];
  const float* bnm  = (const float*)d_in[14];
  const float* bnv  = (const float*)d_in[15];
  const float* wa2  = (const float*)d_in[16];
  const float* ba2  = (const float*)d_in[17];

  const size_t S = (size_t)BATCH * NSP * C_DIM;  // 2,097,152 elements
  char*  ws  = (char*)d_ws;
  size_t off = 0;
  auto alloc = [&](size_t bytes) { char* p = ws + off; off += (bytes + 255) & ~(size_t)255; return p; };
  bf16_t* feaT1 = (bf16_t*)alloc(S * 2);
  bf16_t* feaT2 = (bf16_t*)alloc(S * 2);
  bf16_t* Q1 = (bf16_t*)alloc(S * 2); bf16_t* K1 = (bf16_t*)alloc(S * 2);
  bf16_t* V1 = (bf16_t*)alloc(S * 2);
  bf16_t* Q2 = (bf16_t*)alloc(S * 2); bf16_t* K2 = (bf16_t*)alloc(S * 2);
  bf16_t* V2 = (bf16_t*)alloc(S * 2);
  bf16_t* av1  = (bf16_t*)alloc(S * 2); bf16_t* av2  = (bf16_t*)alloc(S * 2);
  bf16_t* avm1 = (bf16_t*)alloc(S * 2); bf16_t* avm2 = (bf16_t*)alloc(S * 2);
  bf16_t* h1 = (bf16_t*)alloc(S * 4);  bf16_t* h2 = (bf16_t*)alloc(S * 4);
  const int WCC = C_DIM * C_DIM;  // 65536
  bf16_t* wqB  = (bf16_t*)alloc(WCC * 2);
  bf16_t* wkB  = (bf16_t*)alloc(WCC * 2);
  bf16_t* wvB  = (bf16_t*)alloc(WCC * 2);
  bf16_t* wmB  = (bf16_t*)alloc(WCC * 2);
  bf16_t* wa1B = (bf16_t*)alloc(WCC * 8);
  bf16_t* wa2B = (bf16_t*)alloc(WCC * 4);

  // 0) weight pre-convert fp32 -> bf16 (lets GEMM tiles use async-to-LDS)
  k_f32_to_bf16<<<WCC / 256, 256, 0, stream>>>(wq, wqB, WCC);
  k_f32_to_bf16<<<WCC / 256, 256, 0, stream>>>(wk, wkB, WCC);
  k_f32_to_bf16<<<WCC / 256, 256, 0, stream>>>(wv, wvB, WCC);
  k_f32_to_bf16<<<WCC / 256, 256, 0, stream>>>(wm, wmB, WCC);
  k_f32_to_bf16<<<WCC * 4 / 256, 256, 0, stream>>>(wa1, wa1B, WCC * 4);
  k_f32_to_bf16<<<WCC * 2 / 256, 256, 0, stream>>>(wa2, wa2B, WCC * 2);

  // 1) fp32 (B,C,N) -> bf16 (B,N,C)
  k_to_cl_bf16<<<(unsigned)(S / 256), 256, 0, stream>>>(fea1, feaT1);
  k_to_cl_bf16<<<(unsigned)(S / 256), 256, 0, stream>>>(fea2, feaT2);

  // 2) QKV projections (bf16 WMMA GEMMs with head-scatter epilogues)
  dim3 g256(NSP / 128, C_DIM / 128, BATCH);
  k_gemm_cl<3><<<g256, 128, 0, stream>>>(feaT1, nullptr, wqB, bq, nullptr, nullptr, nullptr, nullptr, nullptr, Q1, C_DIM, C_DIM, C_DIM, NSP);
  k_gemm_cl<3><<<g256, 128, 0, stream>>>(feaT1, nullptr, wkB, bk, nullptr, nullptr, nullptr, nullptr, nullptr, K1, C_DIM, C_DIM, C_DIM, NSP);
  k_gemm_cl<4><<<g256, 128, 0, stream>>>(feaT1, nullptr, wvB, bv, nullptr, nullptr, nullptr, nullptr, nullptr, V1, C_DIM, C_DIM, C_DIM, NSP);
  k_gemm_cl<3><<<g256, 128, 0, stream>>>(feaT2, nullptr, wqB, bq, nullptr, nullptr, nullptr, nullptr, nullptr, Q2, C_DIM, C_DIM, C_DIM, NSP);
  k_gemm_cl<3><<<g256, 128, 0, stream>>>(feaT2, nullptr, wkB, bk, nullptr, nullptr, nullptr, nullptr, nullptr, K2, C_DIM, C_DIM, C_DIM, NSP);
  k_gemm_cl<4><<<g256, 128, 0, stream>>>(feaT2, nullptr, wvB, bv, nullptr, nullptr, nullptr, nullptr, nullptr, V2, C_DIM, C_DIM, C_DIM, NSP);

  // 3) cross attention: stream1 attends to stream2 and vice versa
  dim3 ga(NSP / 64, BATCH * HEADS);
  k_flash_attn<<<ga, 128, 0, stream>>>(Q1, K2, V2, av1, NSP, NSP);
  k_flash_attn<<<ga, 128, 0, stream>>>(Q2, K1, V1, av2, NSP, NSP);

  // 4) wm projection
  k_gemm_cl<0><<<g256, 128, 0, stream>>>(av1, nullptr, wmB, bm, nullptr, nullptr, nullptr, nullptr, nullptr, avm1, C_DIM, C_DIM, C_DIM, NSP);
  k_gemm_cl<0><<<g256, 128, 0, stream>>>(av2, nullptr, wmB, bm, nullptr, nullptr, nullptr, nullptr, nullptr, avm2, C_DIM, C_DIM, C_DIM, NSP);

  // 5) wa1 on concat([fea, av]) + BN + ReLU
  dim3 g512(NSP / 128, (2 * C_DIM) / 128, BATCH);
  k_gemm_cl<1><<<g512, 128, 0, stream>>>(feaT1, avm1, wa1B, ba1, bng, bnb, bnm, bnv, nullptr, h1, 2 * C_DIM, C_DIM, 2 * C_DIM, NSP);
  k_gemm_cl<1><<<g512, 128, 0, stream>>>(feaT2, avm2, wa1B, ba1, bng, bnb, bnm, bnv, nullptr, h2, 2 * C_DIM, C_DIM, 2 * C_DIM, NSP);

  // 6) wa2 + residual -> fp32 outputs (out1 | out2 concatenated flat)
  float* out1 = (float*)d_out;
  float* out2 = out1 + S;
  k_gemm_cl<2><<<g256, 128, 0, stream>>>(h1, nullptr, wa2B, ba2, nullptr, nullptr, nullptr, nullptr, fea1, out1, 2 * C_DIM, 2 * C_DIM, C_DIM, NSP);
  k_gemm_cl<2><<<g256, 128, 0, stream>>>(h2, nullptr, wa2B, ba2, nullptr, nullptr, nullptr, nullptr, fea2, out2, 2 * C_DIM, 2 * C_DIM, C_DIM, NSP);
}
